// CrossAttentionBlock_13408887898357
// MI455X (gfx1250) — matrix operand order repaired and checked by reading, not compiled, not run on hardware
//
#include <hip/hip_runtime.h>

#define NQn 16384
#define Dm 256
#define NH 8
#define HD 32
#define PP 5
#define KK 25
#define QT 8
#define ROWS (QT * KK)
#define IMG_H 256
#define IMG_W 256

typedef __attribute__((ext_vector_type(16))) __bf16 bf16x16;
typedef __attribute__((ext_vector_type(8)))  __bf16 bf16x8;
typedef __attribute__((ext_vector_type(8)))  float  f32x8;

union FragU { bf16x16 v; bf16x8 h[2]; __bf16 e[16]; };

static __device__ __forceinline__ __bf16 f2b(float x) { return (__bf16)x; }

// 16 contiguous bf16 (32B), 16B-aligned
static __device__ __forceinline__ bf16x16 ld16(const __bf16* p) {
  FragU u;
  u.h[0] = *(const bf16x8*)(p);
  u.h[1] = *(const bf16x8*)(p + 8);
  return u.v;
}
// A fragment (16x32 bf16) from a bf16 row: lanes<16 hold K {0..7,16..23}, lanes>=16 {8..15,24..31}
static __device__ __forceinline__ bf16x16 ldA(const __bf16* rowp, int kbase, int lane) {
  int o0 = kbase + ((lane & 16) ? 8 : 0);
  FragU u;
  u.h[0] = *(const bf16x8*)(rowp + o0);
  u.h[1] = *(const bf16x8*)(rowp + o0 + 16);
  return u.v;
}
// A fragment from an f32 row (convert on read; 4x 16B LDS loads + pk cvt)
static __device__ __forceinline__ bf16x16 ldA_f32(const float* rowp, int kbase, int lane) {
  int o0 = kbase + ((lane & 16) ? 8 : 0);
  float4 a = *(const float4*)(rowp + o0);
  float4 b = *(const float4*)(rowp + o0 + 4);
  float4 c = *(const float4*)(rowp + o0 + 16);
  float4 d = *(const float4*)(rowp + o0 + 20);
  FragU u;
  u.e[0]=f2b(a.x); u.e[1]=f2b(a.y); u.e[2]=f2b(a.z); u.e[3]=f2b(a.w);
  u.e[4]=f2b(b.x); u.e[5]=f2b(b.y); u.e[6]=f2b(b.z); u.e[7]=f2b(b.w);
  u.e[8]=f2b(c.x); u.e[9]=f2b(c.y); u.e[10]=f2b(c.z); u.e[11]=f2b(c.w);
  u.e[12]=f2b(d.x); u.e[13]=f2b(d.y); u.e[14]=f2b(d.z); u.e[15]=f2b(d.w);
  return u.v;
}
// B fragment (32x16 bf16): lane col fixed; lanes<16 hold K {0..7,16..23}, lanes>=16 {8..15,24..31}
static __device__ __forceinline__ bf16x16 ldB(const __bf16* colrowp, int kbase, int lane) {
  return ldA(colrowp, kbase, lane);
}
static __device__ __forceinline__ f32x8 wmma_bf16(bf16x16 a, bf16x16 b, f32x8 c) {
  return __builtin_amdgcn_wmma_f32_16x16x32_bf16(false, a, false, b, (short)0, c, false, false);
}

// ---------------- prep: weights -> bf16 (WkT transposed), rel_table -> bf16 ----------------
extern "C" __global__ void ca_prep(const float* __restrict__ Wq, const float* __restrict__ Wk,
                                   const float* __restrict__ Wv, const float* __restrict__ Wo,
                                   const float* __restrict__ T,
                                   __bf16* __restrict__ Wq_b, __bf16* __restrict__ WkT_b,
                                   __bf16* __restrict__ Wv_b, __bf16* __restrict__ Wo_b,
                                   __bf16* __restrict__ T_b) {
  int t = blockIdx.x * 256 + threadIdx.x; // 0..65535
  if (t < Dm * Dm) {
    Wq_b[t] = f2b(Wq[t]);
    Wv_b[t] = f2b(Wv[t]);
    Wo_b[t] = f2b(Wo[t]);
    int d = t >> 8, i = t & 255;
    WkT_b[i * Dm + d] = f2b(Wk[t]); // WkT[i][d] = Wk[d][i]
  }
  if (t < PP * PP * Dm) T_b[t] = f2b(T[t]);
}

// ---------------- LDS layout (bytes) ----------------
#define OFF_KEYS   0                     // [QT][KK][Dm] f32  = 204800 (async landing zone)
#define OFF_Q      204800                // [QT][Dm] bf16     = 4096
#define OFF_QW     208896                // [QT][NH][Dm] bf16 = 32768  (reused as kbar)
#define OFF_TQW    241664                // [QT][NH][32] f32  = 8192   (reused as o [QT][Dm] f32)
#define OFF_SC     249856                // [QT][NH][32] f32  = 8192   (first used as qin bf16)
#define OFF_SP     258048                // QT*2 floats
#define SMEM_BYTES 258112

extern "C" __global__ void __launch_bounds__(256)
ca_main(const float* __restrict__ queries, const int* __restrict__ qidx,
        const float* __restrict__ subpix, const float* __restrict__ img,
        const float* __restrict__ bq, const float* __restrict__ bv, const float* __restrict__ bo,
        const __bf16* __restrict__ Wq_b, const __bf16* __restrict__ WkT_b,
        const __bf16* __restrict__ Wv_b, const __bf16* __restrict__ Wo_b,
        const __bf16* __restrict__ T_b, float* __restrict__ out) {
  extern __shared__ char smem[];
  float*  s_keysf = (float*)(smem + OFF_KEYS);
  __bf16* s_q     = (__bf16*)(smem + OFF_Q);
  __bf16* s_qW    = (__bf16*)(smem + OFF_QW);
  float*  s_TqW   = (float*)(smem + OFF_TQW);
  float*  s_sc    = (float*)(smem + OFF_SC);
  __bf16* s_qin   = (__bf16*)(smem + OFF_SC); // overlay (phase 0/1 only)
  float*  s_sp    = (float*)(smem + OFF_SP);

  const int tid = threadIdx.x;
  const int lane = tid & 31;
  const int wave = tid >> 5;
  const int gqb = blockIdx.x * QT;
  const bool lolane = (lane & 16) == 0;

  // ---- P0a: fire-and-forget async gather of the 5x5 key neighborhood into LDS (f32) ----
  // 200 rows x 1KB = 12800 16B chunks; chunk c -> lds off = OFF_KEYS + 16c (row-major [q][kk][256]f32)
  for (int it = 0; it < (ROWS * 64) / 256; ++it) {   // 50 iters, consecutive lanes = consecutive 16B
    int c = it * 256 + tid;
    int row = c >> 6;                  // 0..199
    int q = row / KK, kk = row % KK;
    int gq3 = (gqb + q) * 3;
    int b = qidx[gq3 + 0], y = qidx[gq3 + 1], x = qidx[gq3 + 2];
    int yy = min(max(y + kk / PP - 2, 0), IMG_H - 1);
    int xx = min(max(x + kk % PP - 2, 0), IMG_W - 1);
    unsigned long long ga = (unsigned long long)(uintptr_t)img +
        (((((unsigned long long)b * IMG_H + yy) * IMG_W + xx) * Dm + (unsigned long long)(c & 63) * 4) << 2);
    unsigned ldsoff = (unsigned)(OFF_KEYS + c * 16);
    asm volatile("global_load_async_to_lds_b128 %0, %1, off"
                 :: "v"(ldsoff), "v"(ga) : "memory");
  }

  // ---- P0b: stage subpix + queries (bf16) while the gather is in flight ----
  if (tid < QT * 2) s_sp[tid] = subpix[gqb * 2 + tid];
  {
    int row = tid >> 5;        // 0..7
    int c0 = lane * 8;
    const float* src = queries + (size_t)(gqb + row) * Dm + c0;
    __bf16* dst = s_qin + row * Dm + c0;
#pragma unroll
    for (int j = 0; j < 8; ++j) dst[j] = f2b(src[j]);
  }
  __syncthreads();

  // ---- P1: q = queries @ Wq^T + bq  -> s_q (bf16) ----
  for (int nt = wave * 2; nt < wave * 2 + 2; ++nt) {
    int rowq = (lane & 15) & (QT - 1);
    const __bf16* arow = s_qin + rowq * Dm;
    int ncol = nt * 16 + (lane & 15);
    const __bf16* brow = Wq_b + (size_t)ncol * Dm;
    f32x8 acc = {};
#pragma unroll
    for (int ks = 0; ks < 8; ++ks)
      acc = wmma_bf16(ldA(arow, ks * 32, lane), ldB(brow, ks * 32, lane), acc);
    float bias = bq[ncol];
    if (lolane) {   // rows 0..7 only; lanes>=16 hold padded rows 8..15
#pragma unroll
      for (int r = 0; r < 8; ++r) s_q[r * Dm + ncol] = f2b(acc[r] + bias);
    }
  }
  __syncthreads();

  // ---- P2: qW[n,h,:] = q_h[n,:] @ Wk_h  (K=32 per head) -> s_qW (bf16) ----
  for (int j = wave; j < NH * 16; j += 8) {
    int h = j >> 4, nt = j & 15;
    int rowq = (lane & 15) & (QT - 1);
    const __bf16* arow = s_q + rowq * Dm + h * HD;
    int icol = nt * 16 + (lane & 15);
    const __bf16* brow = WkT_b + (size_t)icol * Dm + h * HD;
    f32x8 acc = {};
    acc = wmma_bf16(ldA(arow, 0, lane), ldB(brow, 0, lane), acc);
    if (lolane) {
#pragma unroll
      for (int r = 0; r < 8; ++r) s_qW[(r * NH + h) * Dm + icol] = f2b(acc[r]);
    }
  }
  __syncthreads();

  // ---- P3: TqW[q,h,t] = T[t,:] . qW[q,h,:] ----
  {
    int mt = wave >> 2, nt = wave & 3;
    int trow = min(mt * 16 + (lane & 15), KK - 1);
    const __bf16* arow = T_b + trow * Dm;
    int col = nt * 16 + (lane & 15);     // 0..63 -> (q,h)
    int cq = col >> 3, ch = col & 7;
    const __bf16* brow = s_qW + (cq * NH + ch) * Dm;
    f32x8 acc = {};
#pragma unroll
    for (int ks = 0; ks < 8; ++ks)
      acc = wmma_bf16(ldA(arow, ks * 32, lane), ldB(brow, ks * 32, lane), acc);
    int rbase = (lane & 16) ? 8 : 0;
#pragma unroll
    for (int r = 0; r < 8; ++r) {
      int tt = mt * 16 + rbase + r;
      if (tt < KK) s_TqW[(cq * NH + ch) * 32 + tt] = acc[r];
    }
  }

  // ---- wait for the async gather, then everyone sees keys ----
  asm volatile("s_wait_asynccnt 0x0" ::: "memory");
  __syncthreads();

  // ---- P4: scores[q,h,kk] = (keys.qW + bilin(TqW)) / sqrt(hd) ----
  {
    int q = wave;
    float suby = s_sp[q * 2 + 0], subx = s_sp[q * 2 + 1];
    int hcol = (lane & 15) & (NH - 1);
    const __bf16* brow = s_qW + (q * NH + hcol) * Dm;
    for (int mt = 0; mt < 2; ++mt) {
      int kkrow = min(mt * 16 + (lane & 15), KK - 1);
      const float* arow = s_keysf + (q * KK + kkrow) * Dm;
      f32x8 acc = {};
#pragma unroll
      for (int ks = 0; ks < 8; ++ks)
        acc = wmma_bf16(ldA_f32(arow, ks * 32, lane), ldB(brow, ks * 32, lane), acc);
      if ((lane & 15) < NH) {
        int h = lane & 15;
        int rbase = (lane & 16) ? 8 : 0;
        const float* tq = s_TqW + (q * NH + h) * 32;
#pragma unroll
        for (int r = 0; r < 8; ++r) {
          int kk = mt * 16 + rbase + r;
          if (kk < KK) {
            int iy = kk / PP, ix = kk % PP;
            float cy = fminf(fmaxf((float)iy - suby, 0.f), 4.f);
            float cx = fminf(fmaxf((float)ix - subx, 0.f), 4.f);
            int y0 = (int)cy, x0 = (int)cx;
            int y1 = min(y0 + 1, 4), x1 = min(x0 + 1, 4);
            float wy = cy - (float)y0, wx = cx - (float)x0;
            float rp = (tq[y0 * PP + x0] * (1.f - wx) + tq[y0 * PP + x1] * wx) * (1.f - wy)
                     + (tq[y1 * PP + x0] * (1.f - wx) + tq[y1 * PP + x1] * wx) * wy;
            s_sc[(q * NH + h) * 32 + kk] = (acc[r] + rp) * 0.17677669529663687f;
          }
        }
      }
    }
  }
  __syncthreads();

  // ---- P5: softmax over kk (25) per (q,h) ----
  if (tid < QT * NH) {
    float* s = s_sc + tid * 32;
    float m = s[0];
    for (int k = 1; k < KK; ++k) m = fmaxf(m, s[k]);
    float sum = 0.f;
    for (int k = 0; k < KK; ++k) { float e = __expf(s[k] - m); s[k] = e; sum += e; }
    float inv = 1.f / sum;
    for (int k = 0; k < KK; ++k) s[k] *= inv;
  }
  __syncthreads();

  // ---- P6: kbar[q,h,:] = attn[q,h,:] @ keys[q,:,:]  (reuse s_qW storage) ----
  __bf16* s_kb = s_qW;
  for (int j = wave; j < QT * 16; j += 8) {
    int q = j >> 4, nt = j & 15;
    FragU ua, ub;
    int rh = (lane & 15) & (NH - 1);
    const float* att = s_sc + (q * NH + rh) * 32;
    int kbA = (lane & 16) ? 8 : 0;
#pragma unroll
    for (int jj = 0; jj < 16; ++jj) {
      int kk = kbA + (jj < 8 ? jj : 8 + jj);
      ua.e[jj] = (kk < KK) ? f2b(att[kk]) : (__bf16)0.0f;
    }
    int icol = nt * 16 + (lane & 15);
    int kbB = (lane & 16) ? 8 : 0;
#pragma unroll
    for (int jj = 0; jj < 16; ++jj) {
      int kk = kbB + (jj < 8 ? jj : 8 + jj);
      ub.e[jj] = (kk < KK) ? f2b(s_keysf[(q * KK + kk) * Dm + icol]) : (__bf16)0.0f;
    }
    f32x8 acc = {};
    acc = wmma_bf16(ua.v, ub.v, acc);
    if (lolane) {   // rows h=0..7
#pragma unroll
      for (int r = 0; r < 8; ++r) s_kb[(q * NH + r) * Dm + icol] = f2b(acc[r]);
    }
  }
  __syncthreads();

  // ---- P7: o[q, h*32+dd] = kbar[q,h,:] . Wv[h*32+dd,:] + bv  (reuse s_TqW as o) ----
  float* s_o = s_TqW;
  for (int j = wave; j < 16; j += 8) {
    int h = j >> 1, nt = j & 1;
    int rowq = (lane & 15) & (QT - 1);
    const __bf16* arow = s_kb + (rowq * NH + h) * Dm;
    int n = h * HD + nt * 16 + (lane & 15);
    const __bf16* brow = Wv_b + (size_t)n * Dm;
    f32x8 acc = {};
#pragma unroll
    for (int ks = 0; ks < 8; ++ks)
      acc = wmma_bf16(ldA(arow, ks * 32, lane), ldB(brow, ks * 32, lane), acc);
    float bias = bv[n];
    if (lolane) {
#pragma unroll
      for (int r = 0; r < 8; ++r) s_o[r * Dm + n] = acc[r] + bias;
    }
  }
  __syncthreads();

  // ---- P8: out = queries + o @ Wo^T + bo ----
  for (int nt = wave; nt < 16; nt += 8) {
    int rowq = (lane & 15) & (QT - 1);
    const float* arow = s_o + rowq * Dm;
    int ncol = nt * 16 + (lane & 15);
    const __bf16* brow = Wo_b + (size_t)ncol * Dm;
    f32x8 acc = {};
#pragma unroll
    for (int ks = 0; ks < 8; ++ks)
      acc = wmma_bf16(ldA_f32(arow, ks * 32, lane), ldB(brow, ks * 32, lane), acc);
    float bias = bo[ncol];
    if (lolane) {
#pragma unroll
      for (int r = 0; r < 8; ++r) {
        size_t gq = (size_t)(gqb + r);
        out[gq * Dm + ncol] = queries[gq * Dm + ncol] + acc[r] + bias;
      }
    }
  }
}

extern "C" void kernel_launch(void* const* d_in, const int* in_sizes, int n_in,
                              void* d_out, int out_size, void* d_ws, size_t ws_size,
                              hipStream_t stream) {
  const float* queries = (const float*)d_in[0];
  const int*   qidx    = (const int*)d_in[1];
  const float* subpix  = (const float*)d_in[2];
  const float* img     = (const float*)d_in[3];
  const float* Wq      = (const float*)d_in[4];
  const float* bq      = (const float*)d_in[5];
  const float* Wk      = (const float*)d_in[6];
  /* bk (d_in[7]) cancels in softmax */
  const float* Wv      = (const float*)d_in[8];
  const float* bv      = (const float*)d_in[9];
  const float* Wo      = (const float*)d_in[10];
  const float* bo      = (const float*)d_in[11];
  const float* T       = (const float*)d_in[12];

  char* ws = (char*)d_ws;
  __bf16* Wq_b  = (__bf16*)(ws + 0);
  __bf16* WkT_b = (__bf16*)(ws + 131072);
  __bf16* Wv_b  = (__bf16*)(ws + 262144);
  __bf16* Wo_b  = (__bf16*)(ws + 393216);
  __bf16* T_b   = (__bf16*)(ws + 524288);

  ca_prep<<<256, 256, 0, stream>>>(Wq, Wk, Wv, Wo, T, Wq_b, WkT_b, Wv_b, Wo_b, T_b);

  (void)hipFuncSetAttribute(reinterpret_cast<const void*>(ca_main),
                            hipFuncAttributeMaxDynamicSharedMemorySize, SMEM_BYTES);
  ca_main<<<NQn / QT, 256, SMEM_BYTES, stream>>>(queries, qidx, subpix, img, bq, bv, bo,
                                                 Wq_b, WkT_b, Wv_b, Wo_b, T_b, (float*)d_out);
}
